// AttentionClassificationHead_30648886624696
// MI455X (gfx1250) — matrix operand order repaired and checked by reading, not compiled, not run on hardware
//
#include <hip/hip_runtime.h>
#include <hip/hip_bf16.h>
#include <math.h>

typedef __attribute__((ext_vector_type(16))) _Float16 v16h;
typedef __attribute__((ext_vector_type(8)))  _Float16 v8h;
typedef __attribute__((ext_vector_type(4)))  _Float16 v4h;
typedef __attribute__((ext_vector_type(8)))  float    v8f;

#define E_DIM 256
#define T_DIM 6
#define H_DIM 4
#define HD 64
#define WIN 16              // windows per block (one 16-row WMMA M tile)
#define ROWS (WIN * T_DIM)  // 96 x-rows staged per block

// ---------------------------------------------------------------------------
// Prep kernel (1 block): fold query into a 16x256 f16 "qk" B-matrix (rows 0..3
// are per-head folded K-projection vectors, rows 4..15 zero), score bias, and
// convert the three GEMM weights to f16 row-major [N_out][K=256] (exactly the
// WMMA B-fragment-friendly layout).
// ---------------------------------------------------------------------------
__global__ void attn_head_prep(const float* __restrict__ query,
                               const float* __restrict__ in_proj_w,
                               const float* __restrict__ in_proj_b,
                               const float* __restrict__ out_w,
                               const float* __restrict__ w1,
                               float* __restrict__ sb,       // [4]
                               _Float16* __restrict__ qk16,  // [16][256]
                               _Float16* __restrict__ Wv16,  // [256][256]
                               _Float16* __restrict__ Ow16,  // [256][256]
                               _Float16* __restrict__ W116)  // [128][256]
{
    __shared__ float qvecS[E_DIM];
    const int tid = threadIdx.x;

    // qvec = query @ Wq^T + bq   (Wq = rows [0,256) of in_proj_w)
    {
        float s = in_proj_b[tid];
        const float* wr = in_proj_w + tid * E_DIM;
        for (int e = 0; e < E_DIM; ++e) s += query[e] * wr[e];
        qvecS[tid] = s;
    }
    __syncthreads();

    // qk16[h][e] = sum_d qvec[h*64+d] * Wk[h*64+d][e]  (Wk = rows [256,512)),
    // rows 4..15 zero-padded.
    for (int o = tid; o < 16 * E_DIM; o += 256) {
        const int n = o >> 8, e = o & 255;
        float s = 0.f;
        if (n < H_DIM)
            for (int d = 0; d < HD; ++d)
                s += qvecS[n * HD + d] * in_proj_w[(E_DIM + n * HD + d) * E_DIM + e];
        qk16[o] = (_Float16)s;
    }
    // sb[h] = q_h . bk_h
    if (tid < H_DIM) {
        float s = 0.f;
        for (int d = 0; d < HD; ++d)
            s += qvecS[tid * HD + d] * in_proj_b[E_DIM + tid * HD + d];
        sb[tid] = s;
    }
    // weight conversions (Wv = rows [512,768) of in_proj_w)
    for (int i = tid; i < E_DIM * E_DIM; i += 256) {
        Wv16[i] = (_Float16)in_proj_w[2 * E_DIM * E_DIM + i];
        Ow16[i] = (_Float16)out_w[i];
    }
    for (int i = tid; i < (E_DIM / 2) * E_DIM; i += 256)
        W116[i] = (_Float16)w1[i];
}

// 16x16 output tile, K=256 accumulation. A: LDS f16 [16][256] row-major.
// B: global f16 [N][256] row-major (row = output col), base at col-block start.
static __device__ __forceinline__ v8f mm_tile256(const _Float16* __restrict__ A,
                                                 const _Float16* __restrict__ B,
                                                 int lane)
{
    const int r  = lane & 15;
    const int hi = lane >> 4;
    const _Float16* ar = A + r * E_DIM;            // A row (M = lane%16)
    const _Float16* br = B + r * E_DIM;            // B col (N = lane%16)
    v8f c = {};
#pragma unroll
    for (int kk = 0; kk < E_DIM; kk += 32) {
        v8h a0 = *(const v8h*)(ar + kk + 8 * hi);
        v8h a1 = *(const v8h*)(ar + kk + 16 + 8 * hi);
        v16h a, b;
        v8h b0 = *(const v8h*)(br + kk + 8 * hi);
        v8h b1 = *(const v8h*)(br + kk + 16 + 8 * hi);
#pragma unroll
        for (int i = 0; i < 8; ++i) { a[i] = a0[i]; a[8 + i] = a1[i]; b[i] = b0[i]; b[8 + i] = b1[i]; }
        c = __builtin_amdgcn_wmma_f32_16x16x32_f16(false, a, false, b,
                                                   (short)0, c, false, false);
    }
    return c;
}

// ---------------------------------------------------------------------------
// Fused kernel: 16 windows per block, 256 threads (8 wave32).
// ---------------------------------------------------------------------------
__global__ void __launch_bounds__(256)
attn_head_main(const float* __restrict__ x,
               const int*   __restrict__ mask,
               const float* __restrict__ in_proj_b,
               const float* __restrict__ out_b,
               const float* __restrict__ ln_g,
               const float* __restrict__ ln_b,
               const float* __restrict__ b1,
               const float* __restrict__ w2,
               const float* __restrict__ b2,
               const float* __restrict__ sb,
               const _Float16* __restrict__ qk16,
               const _Float16* __restrict__ Wv16,
               const _Float16* __restrict__ Ow16,
               const _Float16* __restrict__ W116,
               float* __restrict__ out)
{
    __shared__ __align__(32) _Float16 xs[ROWS * E_DIM]; // 48KB, multi-purpose
    __shared__ float attwS[WIN * H_DIM * T_DIM];
    __shared__ float sbS[H_DIM];
    __shared__ int   validS[WIN];

    // Overlays of xs (sequenced by barriers):
    _Float16* xbarS = xs;                        // [4][16][256] halves 0..16384
    _Float16* attnS = xs + 16384;                // [16][256]    halves 16384..20480
    float*    yS    = (float*)xs;                // [16][256] f32 bytes 0..16384
    _Float16* alnS  = xs + 8192;                 // [16][256]    halves 8192..12288
    _Float16* h1S   = xs + 12288;                // [16][128]    halves 12288..14336

    const int tid  = threadIdx.x;
    const int lane = tid & 31;
    const int wv   = tid >> 5;
    const int gw0  = blockIdx.x * WIN;           // first window of this block

    if (tid < H_DIM) sbS[tid] = sb[tid];
    // S1: stage 96 x-rows as f16 (packed v4h -> single b64 LDS stores)
    for (int i = tid; i < ROWS * (E_DIM / 4); i += 256) {
        const int row = i >> 6, c4 = (i & 63) * 4;
        const float4 v = *(const float4*)(x + (size_t)(gw0 * T_DIM + row) * E_DIM + c4);
        v4h hv;
        hv[0] = (_Float16)v.x; hv[1] = (_Float16)v.y;
        hv[2] = (_Float16)v.z; hv[3] = (_Float16)v.w;
        *(v4h*)(xs + row * E_DIM + c4) = hv;
    }
    __syncthreads();

    // S2: scores via WMMA: X(96x256) @ qk16^T(256x16) -> cols 0..3 are heads.
    // 6 row tiles, one per wave (waves 6,7 idle here).
    if (wv < T_DIM) {
        v8f c = mm_tile256(xs + wv * 16 * E_DIM, qk16, lane);
        const int h = lane & 15;
        if (h < H_DIM) {
#pragma unroll
            for (int r = 0; r < 8; ++r) {
                const int R = wv * 16 + (lane >> 4) * 8 + r;   // x-row 0..95
                const int w = R / T_DIM, t = R % T_DIM;
                const float s = (c[r] + sbS[h]) * 0.125f;      // / sqrt(64)
                attwS[(w * H_DIM + h) * T_DIM + t] =
                    mask[(gw0 + w) * T_DIM + t] ? s : -1.0e9f;
            }
        }
    }
    __syncthreads();

    // softmax over t (64 tasks) + valid flag
    if (tid < WIN * H_DIM) {
        const int w = tid >> 2, h = tid & 3;
        float* sc = attwS + (w * H_DIM + h) * T_DIM;
        float mx = sc[0];
        for (int t = 1; t < T_DIM; ++t) mx = fmaxf(mx, sc[t]);
        float sum = 0.f;
        for (int t = 0; t < T_DIM; ++t) { float e = __expf(sc[t] - mx); sc[t] = e; sum += e; }
        const float inv = 1.f / sum;
        for (int t = 0; t < T_DIM; ++t) sc[t] *= inv;
        if (h == 0) {
            int any = 0;
            for (int t = 0; t < T_DIM; ++t) any |= mask[(gw0 + w) * T_DIM + t];
            validS[w] = any;
        }
    }
    __syncthreads();

    // S3: xbar[h][w][e] = sum_t attw[w][h][t] * x[w*6+t][e]
    // (each x value loaded once, FMA'd into all 4 head accumulators)
    _Float16 hx[64];
#pragma unroll
    for (int w = 0; w < WIN; ++w) {
        float a0 = 0.f, a1 = 0.f, a2 = 0.f, a3 = 0.f;
#pragma unroll
        for (int t = 0; t < T_DIM; ++t) {
            const float xv = (float)xs[(w * T_DIM + t) * E_DIM + tid];
            a0 += attwS[(w * H_DIM + 0) * T_DIM + t] * xv;
            a1 += attwS[(w * H_DIM + 1) * T_DIM + t] * xv;
            a2 += attwS[(w * H_DIM + 2) * T_DIM + t] * xv;
            a3 += attwS[(w * H_DIM + 3) * T_DIM + t] * xv;
        }
        hx[0 * WIN + w] = (_Float16)a0;
        hx[1 * WIN + w] = (_Float16)a1;
        hx[2 * WIN + w] = (_Float16)a2;
        hx[3 * WIN + w] = (_Float16)a3;
    }
    __syncthreads();   // all xs reads done before overlay writes
#pragma unroll
    for (int i = 0; i < 64; ++i) {
        const int h = i >> 4, w = i & 15;
        xbarS[(h * WIN + w) * E_DIM + tid] = hx[i];
    }
    __syncthreads();

    // S4: attn = xbar_h @ Wv_h^T + bv   (per-head 16x256 @ 256x64; 16 tiles)
#pragma unroll
    for (int i = 0; i < 2; ++i) {
        const int t2 = wv * 2 + i;
        const int h = t2 >> 2, nt = t2 & 3;
        const int n0 = h * HD + nt * 16;
        v8f c = mm_tile256(xbarS + h * WIN * E_DIM, Wv16 + n0 * E_DIM, lane);
        const int col = n0 + (lane & 15);
        const float bvv = in_proj_b[2 * E_DIM + col];
#pragma unroll
        for (int r = 0; r < 8; ++r) {
            const int M = (lane >> 4) * 8 + r;
            attnS[M * E_DIM + col] = (_Float16)(c[r] + bvv);
        }
    }
    __syncthreads();

    // S5: y = attn @ out_w^T + out_b   (16x256 @ 256x256; 16 tiles)
#pragma unroll
    for (int i = 0; i < 2; ++i) {
        const int n0 = (wv * 2 + i) * 16;
        v8f c = mm_tile256(attnS, Ow16 + n0 * E_DIM, lane);
        const int col = n0 + (lane & 15);
        const float bo = out_b[col];
#pragma unroll
        for (int r = 0; r < 8; ++r) {
            const int M = (lane >> 4) * 8 + r;
            yS[M * E_DIM + col] = c[r] + bo;
        }
    }
    __syncthreads();

    // S6: LayerNorm, wave-parallel (2 rows per wave, shfl reductions)
#pragma unroll
    for (int rr = 0; rr < 2; ++rr) {
        const int row = wv * 2 + rr;
        const float* yr = yS + row * E_DIM;
        float s = 0.f, s2 = 0.f;
        for (int e = lane; e < E_DIM; e += 32) { const float v = yr[e]; s += v; s2 += v * v; }
#pragma unroll
        for (int m = 16; m; m >>= 1) { s += __shfl_xor(s, m, 32); s2 += __shfl_xor(s2, m, 32); }
        const float mu  = s * (1.f / E_DIM);
        const float var = fmaxf(s2 * (1.f / E_DIM) - mu * mu, 0.f);
        const float inv = rsqrtf(var + 1e-5f);
        for (int e = lane; e < E_DIM; e += 32)
            alnS[row * E_DIM + e] = (_Float16)((yr[e] - mu) * inv * ln_g[e] + ln_b[e]);
    }
    __syncthreads();

    // S7: h = gelu(aln @ w1^T + b1)   (16x256 @ 256x128; 8 tiles, 1 per wave)
    {
        const int n0 = wv * 16;
        v8f c = mm_tile256(alnS, W116 + n0 * E_DIM, lane);
        const int col = n0 + (lane & 15);
        const float bb = b1[col];
#pragma unroll
        for (int r = 0; r < 8; ++r) {
            const int M = (lane >> 4) * 8 + r;
            const float v = c[r] + bb;
            const float g = 0.5f * v * (1.f + erff(v * 0.70710678118f)); // exact GELU
            h1S[M * (E_DIM / 2) + col] = (_Float16)g;
        }
    }
    __syncthreads();

    // S8: logits = h @ w2^T + b2, masked; wave-parallel (2 rows per wave)
#pragma unroll
    for (int rr = 0; rr < 2; ++rr) {
        const int row = wv * 2 + rr;
        float s = 0.f;
        for (int j = lane; j < E_DIM / 2; j += 32)
            s += (float)h1S[row * (E_DIM / 2) + j] * w2[j];
#pragma unroll
        for (int m = 16; m; m >>= 1) s += __shfl_xor(s, m, 32);
        if (lane == 0) out[gw0 + row] = validS[row] ? (s + b2[0]) : 0.f;
    }
}

extern "C" void kernel_launch(void* const* d_in, const int* in_sizes, int n_in,
                              void* d_out, int out_size, void* d_ws, size_t ws_size,
                              hipStream_t stream) {
    const float* x         = (const float*)d_in[0];
    const int*   mask      = (const int*)d_in[1];
    const float* query     = (const float*)d_in[2];
    const float* in_proj_w = (const float*)d_in[3];
    const float* in_proj_b = (const float*)d_in[4];
    const float* out_w     = (const float*)d_in[5];
    const float* out_b     = (const float*)d_in[6];
    const float* ln_g      = (const float*)d_in[7];
    const float* ln_b      = (const float*)d_in[8];
    const float* w1        = (const float*)d_in[9];
    const float* b1        = (const float*)d_in[10];
    const float* w2        = (const float*)d_in[11];
    const float* b2        = (const float*)d_in[12];
    float* out = (float*)d_out;

    char* ws = (char*)d_ws;
    float*    sb   = (float*)ws;                            //  16 B
    _Float16* qk16 = (_Float16*)(ws + 512);                 //   8 KB
    _Float16* Wv16 = (_Float16*)(ws + 8704);                // 128 KB
    _Float16* Ow16 = (_Float16*)(ws + 8704 + 131072);       // 128 KB
    _Float16* W116 = (_Float16*)(ws + 8704 + 262144);       //  64 KB

    attn_head_prep<<<1, 256, 0, stream>>>(query, in_proj_w, in_proj_b, out_w, w1,
                                          sb, qk16, Wv16, Ow16, W116);

    const int n_windows = out_size;          // B*N = 32768
    const int blocks = n_windows / WIN;      // 2048
    attn_head_main<<<blocks, 256, 0, stream>>>(x, mask, in_proj_b, out_b, ln_g, ln_b,
                                               b1, w2, b2, sb, qk16, Wv16, Ow16, W116, out);
}
